// Executor_46334107189311
// MI455X (gfx1250) — matrix operand order repaired and checked by reading, not compiled, not run on hardware
//
#include <hip/hip_runtime.h>
#include <cstddef>
#include <cstdint>

// ---------- types for WMMA ----------
typedef __bf16 bf16_t;
typedef bf16_t v16bf __attribute__((ext_vector_type(16)));
typedef bf16_t v8bf  __attribute__((ext_vector_type(8)));
typedef float  v8f   __attribute__((ext_vector_type(8)));

#define WMMA_BF16(a, b, c) \
  __builtin_amdgcn_wmma_f32_16x16x32_bf16(false, (a), false, (b), (short)0, (c), false, false)

// ---------- problem constants ----------
#define BATCH 4096
#define PDIM  768
#define KSLOT 8
#define SLOT  256
#define HID   256
#define NCODE 512
#define MROWS (BATCH * KSLOT)   // 32768
#define K1DIM (SLOT + PDIM)     // 1024

// =====================================================================
// Per-lane WMMA tile loaders following CDNA5 ISA VGPR layouts (wave32).
// A 16x32 bf16: lanes 0-15 row M=lane, K={0..7,16..23}; lanes 16-31 row
//               M=lane-16, K={8..15,24..31}  -> two aligned 16B loads/lane.
// B 32x16 bf16: lane = column N%16, lanes 0-15 K={0..7,16..23}, lanes 16-31 K={8..15,24..31}
//               -> two aligned 16B loads/lane from [N,K] storage.
// C/D 16x16 f32: N = lane%16, VGPR r -> M = r (lanes 0-15) / r+8 (lanes 16-31).
// =====================================================================

__device__ __forceinline__ v16bf combine8(v8bf lo, v8bf hi) {
  v16bf a;
#pragma unroll
  for (int e = 0; e < 8; ++e) { a[e] = lo[e]; a[e + 8] = hi[e]; }
  return a;
}

// A fragment from row-major bf16 activations.
__device__ __forceinline__ v16bf wmma_load_a_bf16(const bf16_t* A, int lda, int mBase, int k0) {
  int lane = threadIdx.x & 31;
  const bf16_t* p = A + (size_t)(mBase + (lane & 15)) * lda + k0 + ((lane & 16) ? 8 : 0);
  return combine8(*(const v8bf*)p, *(const v8bf*)(p + 16));
}

// B fragment from bf16 weights stored [N, K] row-major (K contiguous).
__device__ __forceinline__ v16bf wmma_load_bT(const bf16_t* T, int ldk, int k0, int nBase) {
  int lane = threadIdx.x & 31;
  const bf16_t* p = T + (size_t)(nBase + (lane & 15)) * ldk + k0 + ((lane & 16) ? 8 : 0);
  __builtin_prefetch((const void*)(p + 32), 0, 1);   // next k-step -> global_prefetch_b8
  return combine8(*(const v8bf*)p, *(const v8bf*)(p + 16));  // K={0..7,16..23}/{8..15,24..31}, 16B aligned
}

__device__ __forceinline__ void wmma_store_f32(float* C, int ldc, int mBase, int nBase, v8f acc) {
  int lane = threadIdx.x & 31;
  int n  = nBase + (lane & 15);
  int m0 = mBase + ((lane & 16) ? 8 : 0);
#pragma unroll
  for (int r = 0; r < 8; ++r) C[(size_t)(m0 + r) * ldc + n] = acc[r];
}

__device__ __forceinline__ void wmma_store_bf16(bf16_t* C, int ldc, int mBase, int nBase, v8f acc) {
  int lane = threadIdx.x & 31;
  int n  = nBase + (lane & 15);
  int m0 = mBase + ((lane & 16) ? 8 : 0);
#pragma unroll
  for (int r = 0; r < 8; ++r) C[(size_t)(m0 + r) * ldc + n] = (bf16_t)acc[r];
}

// =====================================================================
// One-shot prep: fp32 -> bf16 (optionally transposing [K,N] -> [N,K]).
// =====================================================================
__global__ void k_cast(const float* __restrict__ s, bf16_t* __restrict__ d, int n) {
  int i = blockIdx.x * 256 + threadIdx.x;
  if (i < n) d[i] = (bf16_t)s[i];
}
__global__ void k_cast_t(const float* __restrict__ s, bf16_t* __restrict__ d, int K, int N) {
  int i = blockIdx.x * 256 + threadIdx.x;
  if (i < K * N) {
    int n = i / K, k = i % K;            // d is [N, K]
    d[i] = (bf16_t)s[(size_t)k * N + n];
  }
}

// =====================================================================
// Stage 1: Qb = bf16(Xb @ W^T-layout + bias)   (branch = blockIdx.z)
// grid (B/64, P/32, 2), block 32; wave tile 64x32 (4 M x 2 N subtiles)
// =====================================================================
__global__ void k_qproj(const bf16_t* __restrict__ Xb,
                        const bf16_t* __restrict__ WfcT, const bf16_t* __restrict__ WfpT,
                        const float* __restrict__ bfc, const float* __restrict__ bfp,
                        bf16_t* __restrict__ Qcb, bf16_t* __restrict__ Qpb) {
  const bf16_t* WT   = blockIdx.z ? WfpT : WfcT;
  const float*  bias = blockIdx.z ? bfp  : bfc;
  bf16_t*       Q    = blockIdx.z ? Qpb  : Qcb;
  int mBase = blockIdx.x * 64, nBase = blockIdx.y * 32;
  v8f acc[4][2] = {};
  for (int k0 = 0; k0 < PDIM; k0 += 32) {
    v16bf b0 = wmma_load_bT(WT, PDIM, k0, nBase);
    v16bf b1 = wmma_load_bT(WT, PDIM, k0, nBase + 16);
#pragma unroll
    for (int t = 0; t < 4; ++t) {
      v16bf a = wmma_load_a_bf16(Xb, PDIM, mBase + 16 * t, k0);
      acc[t][0] = WMMA_BF16(a, b0, acc[t][0]);
      acc[t][1] = WMMA_BF16(a, b1, acc[t][1]);
    }
  }
  int lane = threadIdx.x & 31;
#pragma unroll
  for (int st = 0; st < 2; ++st) {
    float bv = bias[nBase + st * 16 + (lane & 15)];
#pragma unroll
    for (int t = 0; t < 4; ++t) {
#pragma unroll
      for (int r = 0; r < 8; ++r) acc[t][st][r] += bv;
      wmma_store_bf16(Q, PDIM, mBase + 16 * t, nBase + st * 16, acc[t][st]);
    }
  }
}

// =====================================================================
// Stage 2a: logits = (Q @ Kcb^T) / sqrt(P)   [B, 512] fp32
// grid (B/64, 512/32, 2), block 32
// =====================================================================
__global__ void k_logits(const bf16_t* __restrict__ Qcb, const bf16_t* __restrict__ Qpb,
                         const bf16_t* __restrict__ KcB, const bf16_t* __restrict__ KpB,
                         float* __restrict__ Lc, float* __restrict__ Lp) {
  const bf16_t* Q  = blockIdx.z ? Qpb : Qcb;
  const bf16_t* Kb = blockIdx.z ? KpB : KcB;   // codebooks are [N=512, K=768] already
  float*        L  = blockIdx.z ? Lp  : Lc;
  int mBase = blockIdx.x * 64, nBase = blockIdx.y * 32;
  v8f acc[4][2] = {};
  for (int k0 = 0; k0 < PDIM; k0 += 32) {
    v16bf b0 = wmma_load_bT(Kb, PDIM, k0, nBase);
    v16bf b1 = wmma_load_bT(Kb, PDIM, k0, nBase + 16);
#pragma unroll
    for (int t = 0; t < 4; ++t) {
      v16bf a = wmma_load_a_bf16(Q, PDIM, mBase + 16 * t, k0);
      acc[t][0] = WMMA_BF16(a, b0, acc[t][0]);
      acc[t][1] = WMMA_BF16(a, b1, acc[t][1]);
    }
  }
  const float inv_scale = 0.036084391824351615f;  // 1/sqrt(768)
#pragma unroll
  for (int st = 0; st < 2; ++st)
#pragma unroll
    for (int t = 0; t < 4; ++t) {
#pragma unroll
      for (int r = 0; r < 8; ++r) acc[t][st][r] *= inv_scale;
      wmma_store_f32(L, NCODE, mBase + 16 * t, nBase + st * 16, acc[t][st]);
    }
}

// =====================================================================
// Stage 2b: row softmax over 512; fp32 in, bf16 probs out.
// Row staged into LDS via gfx1250 async global->LDS loads (ASYNCcnt).
// grid (B, 2), block 256.
// =====================================================================
__global__ void k_softmax(const float* __restrict__ Lc, const float* __restrict__ Lp,
                          bf16_t* __restrict__ LcB, bf16_t* __restrict__ LpB) {
  const float* row = (blockIdx.y ? Lp : Lc) + (size_t)blockIdx.x * NCODE;
  bf16_t*      po  = (blockIdx.y ? LpB : LcB) + (size_t)blockIdx.x * NCODE;
  __shared__ float sh[NCODE];
  __shared__ float red[256];
  int t = threadIdx.x;

  // async stage: each thread copies 2 floats (b64) into LDS.
  uint32_t ldsoff = (uint32_t)(uintptr_t)(void*)&sh[2 * t];   // addr[31:0] = LDS offset
  const float* g = row + 2 * t;
  asm volatile("global_load_async_to_lds_b64 %0, %1, off"
               :: "v"(ldsoff), "v"(g) : "memory");
  asm volatile("s_wait_asynccnt 0x0" ::: "memory");
  __syncthreads();

  float x0 = sh[2 * t], x1 = sh[2 * t + 1];
  red[t] = fmaxf(x0, x1);
  __syncthreads();
  for (int s = 128; s > 0; s >>= 1) {
    if (t < s) red[t] = fmaxf(red[t], red[t + s]);
    __syncthreads();
  }
  float m = red[0];
  __syncthreads();
  float e0 = __expf(x0 - m), e1 = __expf(x1 - m);
  red[t] = e0 + e1;
  __syncthreads();
  for (int s = 128; s > 0; s >>= 1) {
    if (t < s) red[t] += red[t + s];
    __syncthreads();
  }
  float inv = 1.0f / red[0];
  po[2 * t]     = (bf16_t)(e0 * inv);
  po[2 * t + 1] = (bf16_t)(e1 * inv);
}

// =====================================================================
// Stage 2c: c = probs @ Vcb  -> bf16 [B, 768].  grid (B/64, P/32, 2), block 32.
// =====================================================================
__global__ void k_attnout(const bf16_t* __restrict__ LcB, const bf16_t* __restrict__ LpB,
                          const bf16_t* __restrict__ VcT, const bf16_t* __restrict__ VpT,
                          bf16_t* __restrict__ Cvb, bf16_t* __restrict__ Pvb) {
  const bf16_t* Pr = blockIdx.z ? LpB : LcB;
  const bf16_t* VT = blockIdx.z ? VpT : VcT;
  bf16_t*       O  = blockIdx.z ? Pvb : Cvb;
  int mBase = blockIdx.x * 64, nBase = blockIdx.y * 32;
  v8f acc[4][2] = {};
  for (int k0 = 0; k0 < NCODE; k0 += 32) {
    v16bf b0 = wmma_load_bT(VT, NCODE, k0, nBase);
    v16bf b1 = wmma_load_bT(VT, NCODE, k0, nBase + 16);
#pragma unroll
    for (int t = 0; t < 4; ++t) {
      v16bf a = wmma_load_a_bf16(Pr, NCODE, mBase + 16 * t, k0);
      acc[t][0] = WMMA_BF16(a, b0, acc[t][0]);
      acc[t][1] = WMMA_BF16(a, b1, acc[t][1]);
    }
  }
#pragma unroll
  for (int st = 0; st < 2; ++st)
#pragma unroll
    for (int t = 0; t < 4; ++t)
      wmma_store_bf16(O, PDIM, mBase + 16 * t, nBase + st * 16, acc[t][st]);
}

// =====================================================================
// Stage 3 (fused update) -- see round 1/2; now with bf16 A-side inputs.
// grid (MROWS/16), block 256 = 8 waves; wave w owns output columns [32w,32w+32).
// =====================================================================

// virtualized concat A-loader (all bf16): k<256 -> slotsB, else -> vec[row/8]
__device__ __forceinline__ v16bf load_concat_a(const bf16_t* __restrict__ slotsB,
                                               const bf16_t* __restrict__ vec,
                                               int mBase, int k0) {
  int lane = threadIdx.x & 31;
  int gm   = mBase + (lane & 15);
  int kk   = k0 + ((lane & 16) ? 8 : 0);
  const bf16_t* p = (k0 < SLOT) ? (slotsB + (size_t)gm * SLOT + kk)
                                : (vec + (size_t)(gm >> 3) * PDIM + (kk - SLOT));
  return combine8(*(const v8bf*)p, *(const v8bf*)(p + 16));
}

__device__ __forceinline__ v16bf load_a_lds(const float (*h)[HID], int k0) {
  int lane = threadIdx.x & 31;
  int r    = lane & 15;
  int kk   = k0 + ((lane & 16) ? 8 : 0);
  v16bf a;
#pragma unroll
  for (int e = 0; e < 8; ++e)  a[e] = (bf16_t)h[r][kk + e];
#pragma unroll
  for (int e = 8; e < 16; ++e) a[e] = (bf16_t)h[r][kk + e + 8];
  return a;
}

__device__ __forceinline__ void store_h_lds(float (*h)[HID], int nBase, v8f acc,
                                            const float* __restrict__ b1) {
  int lane = threadIdx.x & 31;
  int n  = nBase + (lane & 15);
  int m0 = (lane & 16) ? 8 : 0;
  float bv = b1[n];
#pragma unroll
  for (int r = 0; r < 8; ++r) h[m0 + r][n] = acc[r] + bv;
}

__global__ void __launch_bounds__(256)
k_update(const float* __restrict__ slots, const bf16_t* __restrict__ slotsB,
         const bf16_t* __restrict__ cvec, const bf16_t* __restrict__ pvec,
         const bf16_t* __restrict__ pW1T, const float* __restrict__ pb1,
         const float* __restrict__ pg,  const float* __restrict__ pbt,
         const bf16_t* __restrict__ pW2T, const float* __restrict__ pb2,
         const bf16_t* __restrict__ uW1T, const float* __restrict__ ub1,
         const float* __restrict__ ug,  const float* __restrict__ ubt,
         const bf16_t* __restrict__ uW2T, const float* __restrict__ ub2,
         float* __restrict__ out) {
  __shared__ float hP[16][HID];
  __shared__ float hU[16][HID];
  __shared__ float redS[32][8];
  __shared__ float redQ[32][8];
  __shared__ float mu[32];
  __shared__ float rstd[32];

  const int mBase = blockIdx.x * 16;            // row tile in [0, 32768)
  const int wave  = threadIdx.x >> 5;
  const int lane  = threadIdx.x & 31;
  const int nB    = wave * 32;                  // this wave's 32 output columns

  // ---- layer 1: both branches, 2 column-subtiles per wave ----
  v8f aP0 = {}, aP1 = {}, aU0 = {}, aU1 = {};
  for (int k0 = 0; k0 < K1DIM; k0 += 32) {
    v16bf aPres = load_concat_a(slotsB, cvec, mBase, k0);
    v16bf aUp;
    if (k0 < SLOT) aUp = aPres;                 // slots region identical
    else           aUp = load_concat_a(slotsB, pvec, mBase, k0);
    v16bf bP0 = wmma_load_bT(pW1T, K1DIM, k0, nB);
    v16bf bP1 = wmma_load_bT(pW1T, K1DIM, k0, nB + 16);
    v16bf bU0 = wmma_load_bT(uW1T, K1DIM, k0, nB);
    v16bf bU1 = wmma_load_bT(uW1T, K1DIM, k0, nB + 16);
    aP0 = WMMA_BF16(aPres, bP0, aP0);
    aP1 = WMMA_BF16(aPres, bP1, aP1);
    aU0 = WMMA_BF16(aUp,  bU0, aU0);
    aU1 = WMMA_BF16(aUp,  bU1, aU1);
  }
  store_h_lds(hP, nB,      aP0, pb1);
  store_h_lds(hP, nB + 16, aP1, pb1);
  store_h_lds(hU, nB,      aU0, ub1);
  store_h_lds(hU, nB + 16, aU1, ub1);
  __syncthreads();

  // ---- LayerNorm + ReLU over 32 rows (16 pres + 16 up) of 256 ----
  {
    int row = threadIdx.x >> 3;   // 0..31
    int sub = threadIdx.x & 7;    // 8 threads per row, 32 cols each
    float (*h)[HID] = (row < 16) ? hP : hU;
    int r = row & 15;
    float s = 0.f, q = 0.f;
#pragma unroll 4
    for (int c = 0; c < 32; ++c) {
      float v = h[r][sub * 32 + c];
      s += v; q += v * v;
    }
    redS[row][sub] = s; redQ[row][sub] = q;
    __syncthreads();
    if (sub == 0) {
      float ss = 0.f, qq = 0.f;
#pragma unroll
      for (int i = 0; i < 8; ++i) { ss += redS[row][i]; qq += redQ[row][i]; }
      float m = ss * (1.0f / HID);
      float v = qq * (1.0f / HID) - m * m;
      mu[row] = m; rstd[row] = rsqrtf(v + 1e-5f);
    }
    __syncthreads();
    const float* g  = (row < 16) ? pg  : ug;
    const float* bt = (row < 16) ? pbt : ubt;
    float m = mu[row], rs = rstd[row];
#pragma unroll 4
    for (int c = 0; c < 32; ++c) {
      int ci = sub * 32 + c;
      float v = (h[r][ci] - m) * rs * g[ci] + bt[ci];
      h[r][ci] = fmaxf(v, 0.0f);
    }
  }
  __syncthreads();

  // ---- layer 2: both branches from LDS activations ----
  v8f gP0 = {}, gP1 = {}, gU0 = {}, gU1 = {};
  for (int k0 = 0; k0 < HID; k0 += 32) {
    v16bf aP = load_a_lds(hP, k0);
    v16bf aU = load_a_lds(hU, k0);
    v16bf bP0 = wmma_load_bT(pW2T, HID, k0, nB);
    v16bf bP1 = wmma_load_bT(pW2T, HID, k0, nB + 16);
    v16bf bU0 = wmma_load_bT(uW2T, HID, k0, nB);
    v16bf bU1 = wmma_load_bT(uW2T, HID, k0, nB + 16);
    gP0 = WMMA_BF16(aP, bP0, gP0);
    gP1 = WMMA_BF16(aP, bP1, gP1);
    gU0 = WMMA_BF16(aU, bU0, gU0);
    gU1 = WMMA_BF16(aU, bU1, gU1);
  }

  // ---- out = slots + (gate_pres) * (gate_up) ----
  int m0 = mBase + ((lane & 16) ? 8 : 0);
#pragma unroll
  for (int t = 0; t < 2; ++t) {
    int n = nB + t * 16 + (lane & 15);
    float bP = pb2[n], bU = ub2[n];
    v8f gp = t ? gP1 : gP0;
    v8f gu = t ? gU1 : gU0;
#pragma unroll
    for (int r = 0; r < 8; ++r) {
      size_t idx = (size_t)(m0 + r) * SLOT + n;
      out[idx] = slots[idx] + (gp[r] + bP) * (gu[r] + bU);
    }
  }
}

// =====================================================================
// Host-side orchestration
// =====================================================================
extern "C" void kernel_launch(void* const* d_in, const int* in_sizes, int n_in,
                              void* d_out, int out_size, void* d_ws, size_t ws_size,
                              hipStream_t stream) {
  const float* X     = (const float*)d_in[0];   // inst_embed [4096,768]
  const float* slots = (const float*)d_in[1];   // [4096,8,256]
  const float* Wfc   = (const float*)d_in[2];
  const float* bfc   = (const float*)d_in[3];
  const float* Wfp   = (const float*)d_in[4];
  const float* bfp   = (const float*)d_in[5];
  const float* Kc    = (const float*)d_in[6];
  const float* Vc    = (const float*)d_in[7];
  const float* Kp    = (const float*)d_in[8];
  const float* Vp    = (const float*)d_in[9];
  const float* pW1   = (const float*)d_in[10];
  const float* pb1   = (const float*)d_in[11];
  const float* pg    = (const float*)d_in[12];
  const float* pbt   = (const float*)d_in[13];
  const float* pW2   = (const float*)d_in[14];
  const float* pb2   = (const float*)d_in[15];
  const float* uW1   = (const float*)d_in[16];
  const float* ub1   = (const float*)d_in[17];
  const float* ug    = (const float*)d_in[18];
  const float* ubt   = (const float*)d_in[19];
  const float* uW2   = (const float*)d_in[20];
  const float* ub2   = (const float*)d_in[21];
  float* out = (float*)d_out;

  // ---- workspace: fp32 logits, then bf16 region (aligned by construction) ----
  float* ws = (float*)d_ws;
  float* Lc = ws;
  float* Lp = Lc + (size_t)BATCH * NCODE;

  bf16_t* bb = (bf16_t*)(Lp + (size_t)BATCH * NCODE);
  bf16_t* Xb    = bb;  bb += (size_t)BATCH * PDIM;    // [4096,768]
  bf16_t* slotsB= bb;  bb += (size_t)MROWS * SLOT;    // [32768,256]
  bf16_t* Qcb   = bb;  bb += (size_t)BATCH * PDIM;
  bf16_t* Qpb   = bb;  bb += (size_t)BATCH * PDIM;
  bf16_t* LcB   = bb;  bb += (size_t)BATCH * NCODE;   // probs bf16
  bf16_t* LpB   = bb;  bb += (size_t)BATCH * NCODE;
  bf16_t* Cvb   = bb;  bb += (size_t)BATCH * PDIM;
  bf16_t* Pvb   = bb;  bb += (size_t)BATCH * PDIM;
  bf16_t* WfcT  = bb;  bb += (size_t)PDIM * PDIM;     // [768,768] (transposed)
  bf16_t* WfpT  = bb;  bb += (size_t)PDIM * PDIM;
  bf16_t* KcB   = bb;  bb += (size_t)NCODE * PDIM;    // [512,768] (cast only)
  bf16_t* KpB   = bb;  bb += (size_t)NCODE * PDIM;
  bf16_t* VcT   = bb;  bb += (size_t)PDIM * NCODE;    // [768,512]
  bf16_t* VpT   = bb;  bb += (size_t)PDIM * NCODE;
  bf16_t* pW1T  = bb;  bb += (size_t)HID * K1DIM;     // [256,1024]
  bf16_t* uW1T  = bb;  bb += (size_t)HID * K1DIM;
  bf16_t* pW2T  = bb;  bb += (size_t)SLOT * HID;      // [256,256]
  bf16_t* uW2T  = bb;  bb += (size_t)SLOT * HID;

  // ---- prep casts ----
  k_cast  <<<(BATCH * PDIM) / 256, 256, 0, stream>>>(X, Xb, BATCH * PDIM);
  k_cast  <<<(MROWS * SLOT) / 256, 256, 0, stream>>>(slots, slotsB, MROWS * SLOT);
  k_cast_t<<<(PDIM * PDIM) / 256, 256, 0, stream>>>(Wfc, WfcT, PDIM, PDIM);
  k_cast_t<<<(PDIM * PDIM) / 256, 256, 0, stream>>>(Wfp, WfpT, PDIM, PDIM);
  k_cast  <<<(NCODE * PDIM) / 256, 256, 0, stream>>>(Kc, KcB, NCODE * PDIM);
  k_cast  <<<(NCODE * PDIM) / 256, 256, 0, stream>>>(Kp, KpB, NCODE * PDIM);
  k_cast_t<<<(NCODE * PDIM) / 256, 256, 0, stream>>>(Vc, VcT, NCODE, PDIM);
  k_cast_t<<<(NCODE * PDIM) / 256, 256, 0, stream>>>(Vp, VpT, NCODE, PDIM);
  k_cast_t<<<(K1DIM * HID) / 256, 256, 0, stream>>>(pW1, pW1T, K1DIM, HID);
  k_cast_t<<<(K1DIM * HID) / 256, 256, 0, stream>>>(uW1, uW1T, K1DIM, HID);
  k_cast_t<<<(HID * SLOT) / 256, 256, 0, stream>>>(pW2, pW2T, HID, SLOT);
  k_cast_t<<<(HID * SLOT) / 256, 256, 0, stream>>>(uW2, uW2T, HID, SLOT);

  // ---- pipeline ----
  k_qproj  <<<dim3(BATCH / 64, PDIM / 32, 2), 32, 0, stream>>>(Xb, WfcT, WfpT, bfc, bfp, Qcb, Qpb);
  k_logits <<<dim3(BATCH / 64, NCODE / 32, 2), 32, 0, stream>>>(Qcb, Qpb, KcB, KpB, Lc, Lp);
  k_softmax<<<dim3(BATCH, 2), 256, 0, stream>>>(Lc, Lp, LcB, LpB);
  k_attnout<<<dim3(BATCH / 64, PDIM / 32, 2), 32, 0, stream>>>(LcB, LpB, VcT, VpT, Cvb, Pvb);
  k_update <<<dim3(MROWS / 16), 256, 0, stream>>>(slots, slotsB, Cvb, Pvb,
                                                  pW1T, pb1, pg, pbt, pW2T, pb2,
                                                  uW1T, ub1, ug, ubt, uW2T, ub2,
                                                  out);
}
